// MMFP4Expert_27333171871991
// MI455X (gfx1250) — matrix operand order repaired and checked by reading, not compiled, not run on hardware
//
#include <hip/hip_runtime.h>
#include <hip/hip_bf16.h>

typedef __attribute__((ext_vector_type(16))) _Float16 v16h;
typedef __attribute__((ext_vector_type(8)))  _Float16 v8h;
typedef __attribute__((ext_vector_type(8)))  float    v8f;
typedef __attribute__((ext_vector_type(4)))  unsigned v4u;

union H16 { v16h v; v8h h[2]; };

// Problem constants (from reference setup_inputs)
static constexpr int Tt  = 2048;   // tokens
static constexpr int H   = 4096;   // hidden
static constexpr int I   = 14336;  // intermediate
static constexpr int KB  = 64;     // K-step per stage
static constexpr int LDP = 72;     // LDS row stride in halves (64 + 8 pad, 16B-aligned rows)

// ---------------------------------------------------------------------------
// gfx1250 async global->LDS copy of 64 contiguous bytes per lane.
// ISA §15.18: the immediate offset is added to both the LDS and the global address
//             -> one base, immediate offsets.
// Tracked by ASYNCcnt; fence with s_wait_asynccnt before the barrier.
// ---------------------------------------------------------------------------
__device__ inline void async_cp64(_Float16* lds_dst, const _Float16* gsrc) {
    unsigned           lofs = (unsigned)(unsigned long long)(void*)lds_dst;
    unsigned long long ga   = (unsigned long long)(const void*)gsrc;
    asm volatile(
        "global_load_async_to_lds_b128 %0, %1, off\n\t"
        "global_load_async_to_lds_b128 %0, %1, off offset:16\n\t"
        "global_load_async_to_lds_b128 %0, %1, off offset:32\n\t"
        "global_load_async_to_lds_b128 %0, %1, off offset:48"
        :: "v"(lofs), "v"(ga) : "memory");
}
__device__ inline void wait_async0() {
#if __has_builtin(__builtin_amdgcn_s_wait_asynccnt)
    __builtin_amdgcn_s_wait_asynccnt(0);
#else
    asm volatile("s_wait_asynccnt 0x0" ::: "memory");
#endif
}

// FP4 E2M1 nibble -> fp16 bit pattern, fully branchless.
// magnitude code c = nib&7: c>=2 (normal): bits = (c+28)<<9
//                           c<2 (subnormal): bits = c*0x3800 (0 or 0.5)
__device__ inline unsigned fp4h(unsigned nib) {
    unsigned c    = nib & 7u;
    unsigned sgn  = (nib & 8u) << 12;      // sign -> bit 15
    unsigned norm = (c + 28u) << 9;
    unsigned sub  = c * 0x3800u;
    return sgn | ((c >= 2u) ? norm : sub); // value-select -> v_cndmask, no branch
}

// Dequant one packed word (8 nibbles) * scale -> 8 halves to LDS.
// Assemble bit patterns pairwise in u32, bitcast, packed multiply (v_pk_mul_f16).
__device__ inline void dq8(unsigned w, _Float16 scale, _Float16* dst) {
    v4u u;
#pragma unroll
    for (int p = 0; p < 4; ++p) {
        unsigned n0 = (w >> (8 * p))     & 0xFu;
        unsigned n1 = (w >> (8 * p + 4)) & 0xFu;
        u[p] = fp4h(n0) | (fp4h(n1) << 16);
    }
    v8h v  = __builtin_bit_cast(v8h, u);
    v8h sc = { scale, scale, scale, scale, scale, scale, scale, scale };
    *(v8h*)dst = v * sc;
}

// A fragment (16x32 f16) at K-chunk kc: lane<16 -> row=lane, K {0..7},{16..23};
// lane>=16 -> row=lane-16, K {8..15},{24..31}  (relative to kc)
__device__ inline v16h ldA(const _Float16* tile, int mBase, int lane, int kc) {
    int row  = mBase + (lane & 15);
    int koff = kc + ((lane >> 4) & 1) * 8;
    const _Float16* p = tile + row * LDP + koff;
    H16 t;
    t.h[0] = *(const v8h*)(p);
    t.h[1] = *(const v8h*)(p + 16);
    return t.v;
}

// B fragment (32x16 f16) at K-chunk kc: lane<16 -> col=lane, K {0..7},{16..23};
// lane>=16 -> col=lane-16, K {8..15},{24..31}
__device__ inline v16h ldB(const _Float16* tile, int nBase, int lane, int kc) {
    int col  = nBase + (lane & 15);
    int koff = kc + ((lane >> 4) & 1) * 8;
    const _Float16* p = tile + col * LDP + koff;
    H16 t;
    t.h[0] = *(const v8h*)(p);
    t.h[1] = *(const v8h*)(p + 16);
    return t.v;
}

__device__ inline v8f wmma_f16(v16h a, v16h b, v8f c) {
    return __builtin_amdgcn_wmma_f32_16x16x32_f16(
        false, a, false, b, (short)0, c, false, false);
}

// ---------------------------------------------------------------------------
// Kernel 0: one-shot x f32 -> f16 (shared by all 224 I-blocks of kernel 1)
// ---------------------------------------------------------------------------
__global__ void __launch_bounds__(256)
cvt_kernel(const float* __restrict__ x, _Float16* __restrict__ x16) {
    size_t i = ((size_t)blockIdx.x * 256 + threadIdx.x) * 8;
    const float4* p = (const float4*)(x + i);
    float4 f0 = p[0], f1 = p[1];
    v8h h = { (_Float16)f0.x, (_Float16)f0.y, (_Float16)f0.z, (_Float16)f0.w,
              (_Float16)f1.x, (_Float16)f1.y, (_Float16)f1.z, (_Float16)f1.w };
    *(v8h*)(x16 + i) = h;
}

// ---------------------------------------------------------------------------
// Kernel 1: act[T,I] (f16) = SiLU(x @ Wg^T) * (x @ Wu^T), FP4 weights.
// 256 thr = 8 waves; block tile 128(M) x 64(N); wave = 32x32 of gate AND up.
// K-step 64 -> 16 WMMAs per wave per stage.
// ---------------------------------------------------------------------------
__global__ void __launch_bounds__(256)
gateup_kernel(const _Float16* __restrict__ x16,
              const unsigned* __restrict__ gpk, const _Float16* __restrict__ gsc,
              const unsigned* __restrict__ upk, const _Float16* __restrict__ usc,
              _Float16* __restrict__ act) {
    __shared__ alignas(16) _Float16 sX[128 * LDP];
    __shared__ alignas(16) _Float16 sG[64 * LDP];
    __shared__ alignas(16) _Float16 sU[64 * LDP];

    const int bn   = blockIdx.x * 64;
    const int bm   = blockIdx.y * 128;
    const int tid  = threadIdx.x;
    const int lane = tid & 31;
    const int wave = tid >> 5;
    const int wm   = (wave >> 1) * 32;
    const int wn   = (wave & 1) * 32;

    v8f z = {};
    v8f accG[2][2], accU[2][2];
#pragma unroll
    for (int i = 0; i < 2; ++i)
#pragma unroll
        for (int j = 0; j < 2; ++j) { accG[i][j] = z; accU[i][j] = z; }

    const int xr   = tid >> 1;            // 0..127 : x row in tile
    const int xc0  = (tid & 1) * 32;      // 0/32   : half-column base (32 halves/thread)
    const int wnch = tid >> 2;            // 0..63  : weight channel in tile
    const int wkw  = tid & 3;             // 0..3   : first packed word (handles wkw, wkw+4)

    const _Float16* xg = x16 + (size_t)(bm + xr) * H + xc0;
    _Float16*       xl = sX + xr * LDP + xc0;

    for (int k0 = 0; k0 < H; k0 += KB) {
        // ---- activation tile via async global->LDS (64B per thread) ----
        async_cp64(xl, xg + k0);
        // ---- dequant gate/up weight tiles: 64ch x 64k (2 words per matrix) ----
        {
            const size_t wi = (size_t)(bn + wnch) * (H / 8) + (k0 >> 3) + wkw;
            const size_t si = (size_t)(k0 >> 7) * I + bn + wnch;
            _Float16 sg = gsc[si], su = usc[si];
            dq8(gpk[wi],     sg, sG + wnch * LDP + wkw * 8);
            dq8(gpk[wi + 4], sg, sG + wnch * LDP + wkw * 8 + 32);
            dq8(upk[wi],     su, sU + wnch * LDP + wkw * 8);
            dq8(upk[wi + 4], su, sU + wnch * LDP + wkw * 8 + 32);
            if (k0 + KB < H) {  // next K-step packed words -> global_prefetch_b8
                __builtin_prefetch(&gpk[wi + 8], 0, 0);
                __builtin_prefetch(&upk[wi + 8], 0, 0);
            }
        }
        wait_async0();
        __syncthreads();

#pragma unroll
        for (int kc = 0; kc < KB; kc += 32) {
            v16h a0  = ldA(sX, wm,      lane, kc);
            v16h a1  = ldA(sX, wm + 16, lane, kc);
            v16h bg0 = ldB(sG, wn,      lane, kc);
            v16h bg1 = ldB(sG, wn + 16, lane, kc);
            v16h bu0 = ldB(sU, wn,      lane, kc);
            v16h bu1 = ldB(sU, wn + 16, lane, kc);

            accG[0][0] = wmma_f16(a0, bg0, accG[0][0]);
            accG[0][1] = wmma_f16(a0, bg1, accG[0][1]);
            accG[1][0] = wmma_f16(a1, bg0, accG[1][0]);
            accG[1][1] = wmma_f16(a1, bg1, accG[1][1]);
            accU[0][0] = wmma_f16(a0, bu0, accU[0][0]);
            accU[0][1] = wmma_f16(a0, bu1, accU[0][1]);
            accU[1][0] = wmma_f16(a1, bu0, accU[1][0]);
            accU[1][1] = wmma_f16(a1, bu1, accU[1][1]);
        }
        __syncthreads();
    }

    // Epilogue: SiLU(g)*u -> f16. C layout: lane<16: M=r,N=lane ; lane>=16: M=r+8,N=lane-16
    const int lm = ((lane >> 4) & 1) * 8;
    const int ln = lane & 15;
#pragma unroll
    for (int i = 0; i < 2; ++i)
#pragma unroll
        for (int j = 0; j < 2; ++j)
#pragma unroll
            for (int r = 0; r < 8; ++r) {
                int row = bm + wm + i * 16 + lm + r;
                int col = bn + wn + j * 16 + ln;
                float g = accG[i][j][r];
                float u = accU[i][j][r];
                float s = g / (1.0f + __expf(-g));
                act[(size_t)row * I + col] = (_Float16)(s * u);
            }
}

// ---------------------------------------------------------------------------
// Kernel 2: out[T,H] (f32) = act @ Wd^T, FP4 weights, K = I = 14336
// ---------------------------------------------------------------------------
__global__ void __launch_bounds__(256)
down_kernel(const _Float16* __restrict__ act,
            const unsigned* __restrict__ dpk, const _Float16* __restrict__ dsc,
            float* __restrict__ out) {
    __shared__ alignas(16) _Float16 sX[128 * LDP];
    __shared__ alignas(16) _Float16 sW[64 * LDP];

    const int bn   = blockIdx.x * 64;
    const int bm   = blockIdx.y * 128;
    const int tid  = threadIdx.x;
    const int lane = tid & 31;
    const int wave = tid >> 5;
    const int wm   = (wave >> 1) * 32;
    const int wn   = (wave & 1) * 32;

    v8f z = {};
    v8f acc[2][2];
#pragma unroll
    for (int i = 0; i < 2; ++i)
#pragma unroll
        for (int j = 0; j < 2; ++j) acc[i][j] = z;

    const int xr   = tid >> 1;
    const int xc0  = (tid & 1) * 32;
    const int wnch = tid >> 2;
    const int wkw  = tid & 3;

    const _Float16* ag = act + (size_t)(bm + xr) * I + xc0;
    _Float16*       al = sX + xr * LDP + xc0;

    for (int k0 = 0; k0 < I; k0 += KB) {
        async_cp64(al, ag + k0);
        {
            const size_t wi = (size_t)(bn + wnch) * (I / 8) + (k0 >> 3) + wkw;
            const size_t si = (size_t)(k0 >> 7) * H + bn + wnch;
            _Float16 sd = dsc[si];
            dq8(dpk[wi],     sd, sW + wnch * LDP + wkw * 8);
            dq8(dpk[wi + 4], sd, sW + wnch * LDP + wkw * 8 + 32);
            if (k0 + KB < I) __builtin_prefetch(&dpk[wi + 8], 0, 0);
        }
        wait_async0();
        __syncthreads();

#pragma unroll
        for (int kc = 0; kc < KB; kc += 32) {
            v16h a0 = ldA(sX, wm,      lane, kc);
            v16h a1 = ldA(sX, wm + 16, lane, kc);
            v16h b0 = ldB(sW, wn,      lane, kc);
            v16h b1 = ldB(sW, wn + 16, lane, kc);

            acc[0][0] = wmma_f16(a0, b0, acc[0][0]);
            acc[0][1] = wmma_f16(a0, b1, acc[0][1]);
            acc[1][0] = wmma_f16(a1, b0, acc[1][0]);
            acc[1][1] = wmma_f16(a1, b1, acc[1][1]);
        }
        __syncthreads();
    }

    const int lm = ((lane >> 4) & 1) * 8;
    const int ln = lane & 15;
#pragma unroll
    for (int i = 0; i < 2; ++i)
#pragma unroll
        for (int j = 0; j < 2; ++j)
#pragma unroll
            for (int r = 0; r < 8; ++r) {
                int row = bm + wm + i * 16 + lm + r;
                int col = bn + wn + j * 16 + ln;
                out[(size_t)row * H + col] = acc[i][j][r];
            }
}

extern "C" void kernel_launch(void* const* d_in, const int* in_sizes, int n_in,
                              void* d_out, int out_size, void* d_ws, size_t ws_size,
                              hipStream_t stream) {
    (void)in_sizes; (void)n_in; (void)out_size; (void)ws_size;
    const float*    x   = (const float*)d_in[0];
    const unsigned* gpk = (const unsigned*)d_in[1];
    const _Float16* gsc = (const _Float16*)d_in[2];
    const unsigned* upk = (const unsigned*)d_in[3];
    const _Float16* usc = (const _Float16*)d_in[4];
    const unsigned* dpk = (const unsigned*)d_in[5];
    const _Float16* dsc = (const _Float16*)d_in[6];
    // d_in[7] = group_size (always 128 here)

    float*    out  = (float*)d_out;
    _Float16* x16  = (_Float16*)d_ws;                     // [T, H] f16 (16.8 MB)
    _Float16* actw = x16 + (size_t)Tt * H;                // [T, I] f16 (58.7 MB)

    dim3 blk(256);
    cvt_kernel   <<<dim3((Tt * H) / 2048), blk, 0, stream>>>(x, x16);
    gateup_kernel<<<dim3(I / 64, Tt / 128), blk, 0, stream>>>(x16, gpk, gsc, upk, usc, actw);
    down_kernel  <<<dim3(H / 64, Tt / 128), blk, 0, stream>>>(actw, dpk, dsc, out);
}
